// HetConv_46067819217420
// MI455X (gfx1250) — matrix operand order repaired and checked by reading, not compiled, not run on hardware
//
#include <hip/hip_runtime.h>
#include <hip/hip_bf16.h>

// ---------------------------------------------------------------------------
// HetConv (GAT-style heterogeneous conv) for MI455X / gfx1250.
//   H=4 heads, D=32, EF=32, NT=4, ET=8; N=100000 nodes, E=1600000 edges.
// Roofline: edge scatter dominates (~1.6 GB gather/scatter over a working set
// that fits the 192 MB L2) -> optimize for coalesced float4 gathers and
// f32 L2 atomics. The one real GEMM (edge_emb @ W_e^T, K=EF=32) is done with
// v_wmma_f32_16x16x32_f16 (wave32 WMMA, f32 accumulate).
// ---------------------------------------------------------------------------

#define HC_H  4
#define HC_D  32
#define HC_EF 32
#define HC_ET 8
#define HC_HD 128          // H*D
#define NEG_SLOPE 0.2f

typedef __attribute__((ext_vector_type(16))) _Float16 v16h;
typedef __attribute__((ext_vector_type(8)))  float    v8f;

// Order-preserving float<->uint map so global_atomic_max_u32 does float max.
__device__ __forceinline__ unsigned enc_f32(float f) {
    unsigned u = __float_as_uint(f);
    return (u & 0x80000000u) ? ~u : (u | 0x80000000u);
}
__device__ __forceinline__ float dec_f32(unsigned u) {
    return (u & 0x80000000u) ? __uint_as_float(u & 0x7FFFFFFFu)
                             : __uint_as_float(~u);
}

// ---------------------------------------------------------------------------
// K0: ee[et,h] = sum_f ( (edge_emb @ W_e^T)[et, h*EF+f] * attn_e[h,f] )
// One block, 8 waves. Wave `jt` computes output columns [16*jt, 16*jt+16) of
// the 16(padded ET) x 128 product with a single v_wmma_f32_16x16x32_f16
// (K = EF = 32 exactly), scales by attn_e, butterfly-reduces over columns,
// and accumulates into the 32-entry ee table in LDS.
// ---------------------------------------------------------------------------
__global__ void __launch_bounds__(256)
k0_ee_precompute(const float* __restrict__ edge_emb,   // [ET, EF]
                 const float* __restrict__ W_e,        // [H*EF, EF]
                 const float* __restrict__ attn_e,     // [H, EF]
                 float* __restrict__ ee_out)           // [ET, H]
{
    __shared__ float s_ee[HC_ET * HC_H];
    const int tid = threadIdx.x;
    if (tid < HC_ET * HC_H) s_ee[tid] = 0.0f;
    __syncthreads();

    const int jt   = tid >> 5;          // wave id = output n-tile, 0..7
    const int l    = tid & 31;          // lane
    const int mrow = l & 15;            // A row (et) / B col (n) for this lane
    const int kA   = (l < 16) ? 0 : 8;  // 16-bit A layout half-wave K offset
    const int kB   = (l < 16) ? 0 : 16; // unused; B is loaded in A's K order (kA)

    // A: 16x32 f16, rows = edge types (rows >= ET are zero padding).
    // Lane layout (ISA 7.12.2): v0..v3 -> K = kA+{0..7}, v4..v7 -> K = kA+16+{0..7}
    v16h a, b;
#pragma unroll
    for (int v = 0; v < 8; ++v) {
        const int k0 = ((v < 4) ? (2 * v) : (2 * v + 8)) + kA;
        float a0 = 0.0f, a1 = 0.0f;
        if (mrow < HC_ET) {
            a0 = edge_emb[mrow * HC_EF + k0];
            a1 = edge_emb[mrow * HC_EF + k0 + 1];
        }
        a[2 * v]     = (_Float16)a0;
        a[2 * v + 1] = (_Float16)a1;

        // B: 32x16 f16, B[k,n] = W_e[jt*16+n, k]  (i.e. W_e^T slice)
        const int j  = jt * 16 + mrow;        // W_e row / output column
        const int kk = k0;
        b[2 * v]     = (_Float16)W_e[j * HC_EF + kk];
        b[2 * v + 1] = (_Float16)W_e[j * HC_EF + kk + 1];
    }

    v8f c = {};
    c = __builtin_amdgcn_wmma_f32_16x16x32_f16(false, a, false, b,
                                               (short)0, c, false, false);

    // C layout: lane l, vgpr v holds C[m, n] with n = l%16, m = v + 8*(l/16).
    const int col = jt * 16 + mrow;           // global output column (0..127)
    const int h   = col >> 5;                 // head this column belongs to
    const float w = attn_e[h * HC_EF + (col & 31)];

#pragma unroll
    for (int v = 0; v < 8; ++v) {
        float val = c[v] * w;
        // sum over the 16 columns held by this half-wave (masks stay in-half)
        val += __shfl_xor(val, 1, 32);
        val += __shfl_xor(val, 2, 32);
        val += __shfl_xor(val, 4, 32);
        val += __shfl_xor(val, 8, 32);
        const int m = v + ((l < 16) ? 0 : 8); // et row; rows >= ET are padding
        if (mrow == 0 && m < HC_ET)
            atomicAdd(&s_ee[m * HC_H + h], val);
    }
    __syncthreads();
    if (tid < HC_ET * HC_H) ee_out[tid] = s_ee[tid];
}

// ---------------------------------------------------------------------------
// K1: per (node, head): el/er dot products; zero rst; init m (encoded -inf)
// and s. Thread per (n,h), 128B contiguous per thread via float4.
// ---------------------------------------------------------------------------
__global__ void __launch_bounds__(256)
k1_node_prep(const float* __restrict__ feat, const float* __restrict__ fc,
             const float* __restrict__ attn_l, const float* __restrict__ attn_r,
             const int* __restrict__ node_types,
             float* __restrict__ el, float* __restrict__ er,
             unsigned* __restrict__ m_enc, float* __restrict__ s,
             float* __restrict__ rst, int NH)
{
    const int t = blockIdx.x * blockDim.x + threadIdx.x;
    if (t >= NH) return;
    const int n = t / HC_H, h = t % HC_H;
    const int ty = node_types[n];

    const float* fr  = feat + (size_t)n * HC_HD + h * HC_D;
    const float* fcr = fc + ((size_t)ty * HC_H + h) * HC_D;
    const float* al  = attn_l + h * HC_D;
    const float* ar  = attn_r + h * HC_D;
    float* rr = rst + (size_t)n * HC_HD + h * HC_D;

    float sl = 0.0f, sr = 0.0f;
    const float4 z4 = make_float4(0.0f, 0.0f, 0.0f, 0.0f);
#pragma unroll
    for (int d = 0; d < HC_D; d += 4) {
        const float4 f4 = *(const float4*)(fr + d);
        const float4 c4 = *(const float4*)(fcr + d);
        const float4 l4 = *(const float4*)(al + d);
        const float4 r4 = *(const float4*)(ar + d);
        const float fs0 = f4.x * c4.x, fs1 = f4.y * c4.y;
        const float fs2 = f4.z * c4.z, fs3 = f4.w * c4.w;
        sl += fs0 * l4.x + fs1 * l4.y + fs2 * l4.z + fs3 * l4.w;
        sr += fs0 * r4.x + fs1 * r4.y + fs2 * r4.z + fs3 * r4.w;
        *(float4*)(rr + d) = z4;      // rst accumulator starts at zero
    }
    el[t] = sl;
    er[t] = sr;
    m_enc[t] = 0u;                    // encoded "-inf" sentinel
    s[t] = 0.0f;
}

// ---------------------------------------------------------------------------
// K2: per (edge, head): logit + leaky relu -> a-region of d_out; atomic umax
// into segment max (order-preserving encoding). el/er/ee are L2-resident.
// ---------------------------------------------------------------------------
__global__ void __launch_bounds__(256)
k2_edge_score(const int* __restrict__ src, const int* __restrict__ dst,
              const int* __restrict__ e_feat,
              const float* __restrict__ el, const float* __restrict__ er,
              const float* __restrict__ ee,
              float* __restrict__ a_buf, unsigned* __restrict__ m_enc, int EH)
{
    const int t = blockIdx.x * blockDim.x + threadIdx.x;
    if (t >= EH) return;
    const int e = t >> 2, h = t & 3;
    const int sN = src[e], dN = dst[e], et = e_feat[e];
    float x = el[sN * HC_H + h] + er[dN * HC_H + h] + ee[et * HC_H + h];
    x = (x > 0.0f) ? x : NEG_SLOPE * x;
    a_buf[t] = x;                                   // stash logit for K3
    atomicMax(&m_enc[dN * HC_H + h], enc_f32(x));
}

// ---------------------------------------------------------------------------
// K3: heavy pass, one wave per edge. Lane l handles head h=l/8, dims 4*(l%8)..
// ex = exp(e - m[dst]); lane (l%8)==0 adds ex into s and overwrites the logit
// with ex (read-before-write within the wave's in-order instruction stream).
// Message fs[src]*ex scattered with 4 global f32 atomics per lane; the 512B
// feat row gather is fully coalesced across the wave. rst fits in L2.
// ---------------------------------------------------------------------------
__global__ void __launch_bounds__(256)
k3_edge_scatter(const int* __restrict__ src, const int* __restrict__ dst,
                const int* __restrict__ node_types,
                const float* __restrict__ feat, const float* __restrict__ fc,
                const unsigned* __restrict__ m_enc, float* __restrict__ s,
                float* __restrict__ a_buf, float* __restrict__ rst, int E)
{
    const int t = blockIdx.x * blockDim.x + threadIdx.x;
    const int e = t >> 5;
    if (e >= E) return;
    const int l  = t & 31;
    const int h  = l >> 3;
    const int d4 = (l & 7) << 2;

    const int sN = src[e], dN = dst[e];
    const float ev = a_buf[e * HC_H + h];                 // logit from K2
    const float mv = dec_f32(m_enc[dN * HC_H + h]);
    const float ex = __expf(ev - mv);

    if ((l & 7) == 0) {                                   // once per (e,h)
        atomicAdd(&s[dN * HC_H + h], ex);
        a_buf[e * HC_H + h] = ex;                         // stash ex for K5
    }

    const int ty = node_types[sN];
    const float4 f4 = *(const float4*)(feat + (size_t)sN * HC_HD + h * HC_D + d4);
    const float4 c4 = *(const float4*)(fc + (size_t)ty * HC_HD + h * HC_D + d4);
    float* rp = rst + (size_t)dN * HC_HD + h * HC_D + d4;
    atomicAdd(rp + 0, f4.x * c4.x * ex);
    atomicAdd(rp + 1, f4.y * c4.y * ex);
    atomicAdd(rp + 2, f4.z * c4.z * ex);
    atomicAdd(rp + 3, f4.w * c4.w * ex);
}

// ---------------------------------------------------------------------------
// K4: per (node, head): rst = rst / s + feat  (residual). Nodes with no
// in-edges have s==0 and accumulated 0 -> keep just the residual (matches
// the reference, which never divides at node granularity).
// ---------------------------------------------------------------------------
__global__ void __launch_bounds__(256)
k4_node_finalize(const float* __restrict__ feat, const float* __restrict__ s,
                 float* __restrict__ rst, int NH)
{
    const int t = blockIdx.x * blockDim.x + threadIdx.x;
    if (t >= NH) return;
    const int n = t / HC_H, h = t % HC_H;
    const float sv  = s[t];
    const float inv = (sv > 0.0f) ? (1.0f / sv) : 0.0f;
    float* rr = rst + (size_t)n * HC_HD + h * HC_D;
    const float* fr = feat + (size_t)n * HC_HD + h * HC_D;
#pragma unroll
    for (int d = 0; d < HC_D; d += 4) {
        float4 r = *(float4*)(rr + d);
        const float4 f = *(const float4*)(fr + d);
        r.x = r.x * inv + f.x;
        r.y = r.y * inv + f.y;
        r.z = r.z * inv + f.z;
        r.w = r.w * inv + f.w;
        *(float4*)(rr + d) = r;
    }
}

// ---------------------------------------------------------------------------
// K5: per (edge, head): a = ex / s[dst]  (s is final; every edge's dst has
// at least this edge, so s > 0 here).
// ---------------------------------------------------------------------------
__global__ void __launch_bounds__(256)
k5_edge_norm(const int* __restrict__ dst, const float* __restrict__ s,
             float* __restrict__ a_buf, int EH)
{
    const int t = blockIdx.x * blockDim.x + threadIdx.x;
    if (t >= EH) return;
    const int e = t >> 2, h = t & 3;
    a_buf[t] = a_buf[t] / s[dst[e] * HC_H + h];
}

// ---------------------------------------------------------------------------
extern "C" void kernel_launch(void* const* d_in, const int* in_sizes, int n_in,
                              void* d_out, int out_size, void* d_ws, size_t ws_size,
                              hipStream_t stream)
{
    const float* feat       = (const float*)d_in[0];
    const float* fc         = (const float*)d_in[1];
    const float* edge_emb   = (const float*)d_in[2];
    const float* W_e        = (const float*)d_in[3];
    const float* attn_l     = (const float*)d_in[4];
    const float* attn_r     = (const float*)d_in[5];
    const float* attn_e     = (const float*)d_in[6];
    const int*   node_types = (const int*)d_in[7];
    const int*   e_feat     = (const int*)d_in[8];
    const int*   src        = (const int*)d_in[9];
    const int*   dst        = (const int*)d_in[10];

    const int N  = in_sizes[0] / HC_HD;
    const int E  = in_sizes[9];
    const int NH = N * HC_H;
    const int EH = E * HC_H;

    // Output layout: rst [N,H,D] followed by a [E,H,1].
    float* rst   = (float*)d_out;
    float* a_buf = rst + (size_t)N * HC_HD;

    // Workspace: el, er, m_enc, s (N*H each) + ee table (ET*H). ~6.4 MB.
    float*    el    = (float*)d_ws;
    float*    er    = el + NH;
    unsigned* m_enc = (unsigned*)(er + NH);
    float*    s     = (float*)(m_enc + NH);
    float*    ee    = s + NH;

    const int B = 256;
    hipLaunchKernelGGL(k0_ee_precompute, dim3(1), dim3(B), 0, stream,
                       edge_emb, W_e, attn_e, ee);
    hipLaunchKernelGGL(k1_node_prep, dim3((NH + B - 1) / B), dim3(B), 0, stream,
                       feat, fc, attn_l, attn_r, node_types, el, er, m_enc, s, rst, NH);
    hipLaunchKernelGGL(k2_edge_score, dim3((EH + B - 1) / B), dim3(B), 0, stream,
                       src, dst, e_feat, el, er, ee, a_buf, m_enc, EH);
    hipLaunchKernelGGL(k3_edge_scatter, dim3((E * 32 + B - 1) / B), dim3(B), 0, stream,
                       src, dst, node_types, feat, fc, m_enc, s, a_buf, rst, E);
    hipLaunchKernelGGL(k4_node_finalize, dim3((NH + B - 1) / B), dim3(B), 0, stream,
                       feat, s, rst, NH);
    hipLaunchKernelGGL(k5_edge_norm, dim3((EH + B - 1) / B), dim3(B), 0, stream,
                       dst, s, a_buf, EH);
}
